// QRNNProjection_65627100282893
// MI455X (gfx1250) — matrix operand order repaired and checked by reading, not compiled, not run on hardware
//
#include <hip/hip_runtime.h>
#include <hip/hip_bf16.h>

// ---------------------------------------------------------------------------
// QRNN-with-projection for MI455X (gfx1250), wave32 + WMMA f16 (f32 accum).
//   S=2048, B=32, D_IN=512, H=1024, PROJ=512, L=2
// Pipeline per layer:
//   gate GEMM (WMMA, software-pipelined) -> activations (NT stores) ->
//   sequential forget-mult scan (NT loads, 2-way ILP) ->
//   projection GEMM (WMMA) -> LayerNorm (wave32 shfl reductions)
// Cache policy: single-use >L2-sized planes stream NT; reused f16
// activations/weights stay resident in the 192MB L2.
// ---------------------------------------------------------------------------

typedef __attribute__((ext_vector_type(16))) _Float16 v16h;
typedef __attribute__((ext_vector_type(8)))  float    v8f;
typedef __attribute__((ext_vector_type(4)))  float    vf4;
typedef __attribute__((ext_vector_type(4)))  _Float16 vh4;

#define S_LEN  2048
#define B_SZ   32
#define DIN    512
#define H_SZ   1024
#define PROJD  512
#define M_ROWS (S_LEN * B_SZ)     // 65536 GEMM rows
#define NGATE  (3 * H_SZ)         // 3072
#define LN_EPS 1e-5f

#define WMMA_F16(a, b, c) \
  __builtin_amdgcn_wmma_f32_16x16x32_f16(false, (a), false, (b), (short)0, (c), false, false)

// ---------------------------------------------------------------------------
// WMMA fragment gathers (CDNA5 ISA 7.12.2 layouts, wave32).
// A (16xK=32, 16-bit): lane L -> row m=L&15; VGPR0-3 hold K = kh..kh+7
// (kh = 8*(L>>4)), VGPR4-7 hold K = kh+16..kh+23.  Both chunks are 16B
// contiguous in a K-major source -> two b128 loads.
// B (K=32x16, 16-bit): lane L -> col n=L&15; VGPR0-3 hold K = kb..kb+7
// (kb = 8*(L>>4)), VGPR4-7 hold K = kb+16..kb+23 (same as A).
// ---------------------------------------------------------------------------
static __device__ __forceinline__ v16h load_frag_a(const _Float16* tile, int ld) {
  int lane = threadIdx.x & 31;
  int m  = lane & 15;
  int kh = (lane >> 4) << 3;                 // 0 or 8
  const _Float16* p = tile + (size_t)m * ld + kh;
  union { v16h h; vf4 f[2]; } u;
  u.f[0] = *(const vf4*)(p);                 // K = kh .. kh+7
  u.f[1] = *(const vf4*)(p + 16);            // K = kh+16 .. kh+23
  return u.h;
}

static __device__ __forceinline__ v16h load_frag_b(const _Float16* tile, int ld) {
  int lane = threadIdx.x & 31;
  int n  = lane & 15;
  int kb = (lane >> 4) << 3;                 // 0 or 8
  const _Float16* p = tile + (size_t)n * ld + kb;
  union { v16h h; vf4 f[2]; } u;
  u.f[0] = *(const vf4*)(p);                 // K = kb .. kb+7
  u.f[1] = *(const vf4*)(p + 16);            // K = kb+16 .. kb+23
  return u.h;
}

// ---------------------------------------------------------------------------
// fp32 -> f16 staging (vectorized x4); src is read exactly once -> NT load
// ---------------------------------------------------------------------------
__global__ __launch_bounds__(256) void cvt_f32_to_f16_kernel(
    const float* __restrict__ src, _Float16* __restrict__ dst, int n4) {
  int i = blockIdx.x * 256 + threadIdx.x;
  if (i < n4) {
    vf4 v = __builtin_nontemporal_load((const vf4*)src + i);
    vh4 o;
    o[0] = (_Float16)v[0]; o[1] = (_Float16)v[1];
    o[2] = (_Float16)v[2]; o[3] = (_Float16)v[3];
    ((vh4*)dst)[i] = o;                       // reused by GEMMs: keep RT
  }
}

// ---------------------------------------------------------------------------
// Gate GEMM: Y(M,3072) = A(M,512) x W(3072,512)^T + bias, then activations.
// Block = 256 thr = 8 waves arranged 2(M) x 4(N); wave tile = 16x64.
// k-loop is software pipelined: tile k+1 fragments are issued before the
// WMMAs of tile k so the hardware only waits on the older loads.
// Gate planes are written NT (768MB total, read once by the scan).
// ---------------------------------------------------------------------------
__global__ __launch_bounds__(256) void gate_gemm_kernel(
    const _Float16* __restrict__ A,      // M x 512 (K-major)
    const _Float16* __restrict__ W,      // 3072 x 512 (K-major)
    const float*    __restrict__ bias,   // 3072
    float* __restrict__ Zt, float* __restrict__ Fg, float* __restrict__ Og) {
  int wave = threadIdx.x >> 5;
  int lane = threadIdx.x & 31;
  int row0 = blockIdx.y * 32 + (wave & 1) * 16;
  int col0 = blockIdx.x * 256 + (wave >> 1) * 64;

  const _Float16* arow = A + (size_t)row0 * DIN;
  const _Float16* w0 = W + (size_t)(col0 +  0) * DIN;
  const _Float16* w1 = W + (size_t)(col0 + 16) * DIN;
  const _Float16* w2 = W + (size_t)(col0 + 32) * DIN;
  const _Float16* w3 = W + (size_t)(col0 + 48) * DIN;

  v8f acc[4] = {v8f{}, v8f{}, v8f{}, v8f{}};

  // prologue: tile k=0 in flight
  v16h a  = load_frag_a(arow, DIN);
  v16h b0 = load_frag_b(w0, DIN);
  v16h b1 = load_frag_b(w1, DIN);
  v16h b2 = load_frag_b(w2, DIN);
  v16h b3 = load_frag_b(w3, DIN);

  for (int k = 32; k < DIN; k += 32) {
    __builtin_prefetch(arow + ((k + 64) & (DIN - 1)), 0, 3);
    v16h an  = load_frag_a(arow + k, DIN);
    v16h bn0 = load_frag_b(w0 + k, DIN);
    v16h bn1 = load_frag_b(w1 + k, DIN);
    v16h bn2 = load_frag_b(w2 + k, DIN);
    v16h bn3 = load_frag_b(w3 + k, DIN);
    acc[0] = WMMA_F16(a, b0, acc[0]);
    acc[1] = WMMA_F16(a, b1, acc[1]);
    acc[2] = WMMA_F16(a, b2, acc[2]);
    acc[3] = WMMA_F16(a, b3, acc[3]);
    a = an; b0 = bn0; b1 = bn1; b2 = bn2; b3 = bn3;
  }
  acc[0] = WMMA_F16(a, b0, acc[0]);
  acc[1] = WMMA_F16(a, b1, acc[1]);
  acc[2] = WMMA_F16(a, b2, acc[2]);
  acc[3] = WMMA_F16(a, b3, acc[3]);

  int region = col0 >> 10;  // 0=Z(tanh) 1=F(sigmoid) 2=O(sigmoid); block-uniform
  float* plane = (region == 0) ? Zt : ((region == 1) ? Fg : Og);
  int mhi = (lane >> 4) << 3;
  #pragma unroll
  for (int t = 0; t < 4; ++t) {
    int n = col0 + t * 16 + (lane & 15);
    float bv = bias[n];
    int h = n & (H_SZ - 1);
    #pragma unroll
    for (int r = 0; r < 8; ++r) {
      int m = row0 + mhi + r;
      float v = acc[t][r] + bv;
      if (region == 0) {
        // tanh(x) = 1 - 2/(exp(2x)+1)  (saturates correctly at +/-inf)
        v = 1.0f - 2.0f / (__expf(2.0f * v) + 1.0f);
      } else {
        v = 1.0f / (1.0f + __expf(-v));
      }
      __builtin_nontemporal_store(v, plane + (size_t)m * H_SZ + h);
    }
  }
}

// ---------------------------------------------------------------------------
// Forget-mult scan: c_s = f_s*z_s + (1-f_s)*c_{s-1}; Hout = sig(o)*c (f16).
// Each thread owns TWO (b,h) columns (independent chains -> 2x MLP on the
// serial recurrence). Gate planes stream NT (read once, bigger than L2).
// ---------------------------------------------------------------------------
__global__ __launch_bounds__(256) void scan_kernel(
    const float* __restrict__ Zt, const float* __restrict__ Fg,
    const float* __restrict__ Og, _Float16* __restrict__ Hh,
    float* __restrict__ hidden_out) {
  int tid = blockIdx.x * 256 + threadIdx.x;      // 0 .. B*H/2-1
  int b = tid >> 9;                              // 512 threads per batch row
  int h = tid & 511;                             // columns h and h+512
  size_t off0 = (size_t)b * H_SZ + h;
  size_t off1 = off0 + 512;
  const size_t stride = (size_t)B_SZ * H_SZ;
  float c0 = 0.0f, c1 = 0.0f;
  for (int s = 0; s < S_LEN; ++s) {
    float z0 = __builtin_nontemporal_load(Zt + off0);
    float z1 = __builtin_nontemporal_load(Zt + off1);
    float f0 = __builtin_nontemporal_load(Fg + off0);
    float f1 = __builtin_nontemporal_load(Fg + off1);
    float o0 = __builtin_nontemporal_load(Og + off0);
    float o1 = __builtin_nontemporal_load(Og + off1);
    c0 = fmaf(f0, z0 - c0, c0);                  // f*z + (1-f)*c
    c1 = fmaf(f1, z1 - c1, c1);
    Hh[off0] = (_Float16)(o0 * c0);              // consumed by proj GEMM: RT
    Hh[off1] = (_Float16)(o1 * c1);
    off0 += stride;
    off1 += stride;
  }
  hidden_out[(size_t)b * H_SZ + h]       = c0;   // C[S-1], layout (B,H)
  hidden_out[(size_t)b * H_SZ + h + 512] = c1;
}

// ---------------------------------------------------------------------------
// Projection GEMM: Pout(M,512) = Hout(M,1024) x P(512,1024)^T  (no bias)
// Same pipelined WMMA skeleton; output plane is single-use -> NT stores.
// ---------------------------------------------------------------------------
__global__ __launch_bounds__(256) void proj_gemm_kernel(
    const _Float16* __restrict__ A,      // M x 1024 (K-major)
    const _Float16* __restrict__ Pw,     // 512 x 1024 (K-major)
    float* __restrict__ out) {           // M x 512
  int wave = threadIdx.x >> 5;
  int lane = threadIdx.x & 31;
  int row0 = blockIdx.y * 32 + (wave & 1) * 16;
  int col0 = blockIdx.x * 256 + (wave >> 1) * 64;

  const _Float16* arow = A + (size_t)row0 * H_SZ;
  const _Float16* w0 = Pw + (size_t)(col0 +  0) * H_SZ;
  const _Float16* w1 = Pw + (size_t)(col0 + 16) * H_SZ;
  const _Float16* w2 = Pw + (size_t)(col0 + 32) * H_SZ;
  const _Float16* w3 = Pw + (size_t)(col0 + 48) * H_SZ;

  v8f acc[4] = {v8f{}, v8f{}, v8f{}, v8f{}};

  v16h a  = load_frag_a(arow, H_SZ);
  v16h b0 = load_frag_b(w0, H_SZ);
  v16h b1 = load_frag_b(w1, H_SZ);
  v16h b2 = load_frag_b(w2, H_SZ);
  v16h b3 = load_frag_b(w3, H_SZ);

  for (int k = 32; k < H_SZ; k += 32) {
    __builtin_prefetch(arow + ((k + 64) & (H_SZ - 1)), 0, 3);
    v16h an  = load_frag_a(arow + k, H_SZ);
    v16h bn0 = load_frag_b(w0 + k, H_SZ);
    v16h bn1 = load_frag_b(w1 + k, H_SZ);
    v16h bn2 = load_frag_b(w2 + k, H_SZ);
    v16h bn3 = load_frag_b(w3 + k, H_SZ);
    acc[0] = WMMA_F16(a, b0, acc[0]);
    acc[1] = WMMA_F16(a, b1, acc[1]);
    acc[2] = WMMA_F16(a, b2, acc[2]);
    acc[3] = WMMA_F16(a, b3, acc[3]);
    a = an; b0 = bn0; b1 = bn1; b2 = bn2; b3 = bn3;
  }
  acc[0] = WMMA_F16(a, b0, acc[0]);
  acc[1] = WMMA_F16(a, b1, acc[1]);
  acc[2] = WMMA_F16(a, b2, acc[2]);
  acc[3] = WMMA_F16(a, b3, acc[3]);

  int mhi = (lane >> 4) << 3;
  #pragma unroll
  for (int t = 0; t < 4; ++t) {
    int n = col0 + t * 16 + (lane & 15);
    #pragma unroll
    for (int r = 0; r < 8; ++r) {
      int m = row0 + mhi + r;
      __builtin_nontemporal_store(acc[t][r], out + (size_t)m * PROJD + n);
    }
  }
}

// ---------------------------------------------------------------------------
// LayerNorm over PROJ=512 cols. One wave (32 lanes) per row, 16 cols/lane.
// Writes fp32 (if dstf; final output -> NT) and/or f16 (if dsth, feeds the
// next layer's GEMM -> RT, stays in L2).
// ---------------------------------------------------------------------------
__global__ __launch_bounds__(256) void layernorm_kernel(
    const float* __restrict__ src,      // M x 512 (single-use -> NT loads)
    const float* __restrict__ g,        // 512
    const float* __restrict__ be,       // 512
    float* __restrict__ dstf,           // M x 512 or null
    _Float16* __restrict__ dsth) {      // M x 512 or null
  int wave = threadIdx.x >> 5;
  int lane = threadIdx.x & 31;
  int row = blockIdx.x * 8 + wave;
  const float* r = src + (size_t)row * PROJD;

  float v[16];
  float s = 0.0f;
  #pragma unroll
  for (int j = 0; j < 16; ++j) {
    v[j] = __builtin_nontemporal_load(r + lane + 32 * j);
    s += v[j];
  }
  #pragma unroll
  for (int o = 16; o >= 1; o >>= 1) s += __shfl_xor(s, o, 32);
  float mu = s * (1.0f / PROJD);

  float vs = 0.0f;
  #pragma unroll
  for (int j = 0; j < 16; ++j) {
    float d = v[j] - mu;
    vs += d * d;
  }
  #pragma unroll
  for (int o = 16; o >= 1; o >>= 1) vs += __shfl_xor(vs, o, 32);
  float inv = rsqrtf(vs * (1.0f / PROJD) + LN_EPS);

  #pragma unroll
  for (int j = 0; j < 16; ++j) {
    int p = lane + 32 * j;
    float o = (v[j] - mu) * inv * g[p] + be[p];
    size_t idx = (size_t)row * PROJD + p;
    if (dstf) __builtin_nontemporal_store(o, dstf + idx);
    if (dsth) dsth[idx] = (_Float16)o;
  }
}

// ---------------------------------------------------------------------------
// Host orchestration (graph-capture safe: launches only, scratch from d_ws)
// ---------------------------------------------------------------------------
extern "C" void kernel_launch(void* const* d_in, const int* in_sizes, int n_in,
                              void* d_out, int out_size, void* d_ws, size_t ws_size,
                              hipStream_t stream) {
  const float* X     = (const float*)d_in[0];  // (S,B,512)
  const float* W     = (const float*)d_in[1];  // (2,3072,512)
  const float* bias  = (const float*)d_in[2];  // (2,3072)
  const float* P     = (const float*)d_in[3];  // (2,512,1024)
  const float* gamma = (const float*)d_in[4];  // (2,512)
  const float* beta  = (const float*)d_in[5];  // (2,512)

  float* out_main   = (float*)d_out;                        // (S,B,512)
  float* out_hidden = out_main + (size_t)M_ROWS * PROJD;    // (2,B,1024)

  // Workspace carve-out (~1.1 GB total)
  char* ws = (char*)d_ws;
  size_t off = 0;
  auto carve = [&](size_t bytes) -> char* {
    char* p = ws + off;
    off += (bytes + 255) & ~(size_t)255;
    return p;
  };
  _Float16* bf_in = (_Float16*)carve((size_t)M_ROWS * DIN  * 2);   //  64 MB
  _Float16* bf_h  = (_Float16*)carve((size_t)M_ROWS * H_SZ * 2);   // 128 MB
  float* planeZ   = (float*)   carve((size_t)M_ROWS * H_SZ * 4);   // 256 MB
  float* planeF   = (float*)   carve((size_t)M_ROWS * H_SZ * 4);   // 256 MB
  float* planeO   = (float*)   carve((size_t)M_ROWS * H_SZ * 4);   // 256 MB
  float* planeP   = (float*)   carve((size_t)M_ROWS * PROJD * 4);  // 128 MB
  _Float16* Wf    = (_Float16*)carve((size_t)2 * NGATE * DIN  * 2);//   6 MB
  _Float16* Pf    = (_Float16*)carve((size_t)2 * PROJD * H_SZ * 2);//   2 MB
  (void)ws_size; (void)in_sizes; (void)n_in; (void)out_size;

  // One-time f16 staging of inputs/weights
  {
    int n4 = (M_ROWS * DIN) / 4;
    cvt_f32_to_f16_kernel<<<(n4 + 255) / 256, 256, 0, stream>>>(X, bf_in, n4);
  }
  {
    int n4 = (2 * NGATE * DIN) / 4;
    cvt_f32_to_f16_kernel<<<(n4 + 255) / 256, 256, 0, stream>>>(W, Wf, n4);
  }
  {
    int n4 = (2 * PROJD * H_SZ) / 4;
    cvt_f32_to_f16_kernel<<<(n4 + 255) / 256, 256, 0, stream>>>(P, Pf, n4);
  }

  dim3 ggrid(NGATE / 256, M_ROWS / 32);   // (12, 2048)
  dim3 pgrid(PROJD / 256, M_ROWS / 32);   // (2, 2048)

  for (int l = 0; l < 2; ++l) {
    gate_gemm_kernel<<<ggrid, 256, 0, stream>>>(
        bf_in, Wf + (size_t)l * NGATE * DIN, bias + l * NGATE,
        planeZ, planeF, planeO);

    scan_kernel<<<(B_SZ * H_SZ / 2) / 256, 256, 0, stream>>>(
        planeZ, planeF, planeO, bf_h, out_hidden + (size_t)l * B_SZ * H_SZ);

    proj_gemm_kernel<<<pgrid, 256, 0, stream>>>(
        bf_h, Pf + (size_t)l * PROJD * H_SZ, planeP);

    layernorm_kernel<<<M_ROWS / 8, 256, 0, stream>>>(
        planeP, gamma + l * PROJD, beta + l * PROJD,
        (l == 1) ? out_main : (float*)nullptr,
        (l == 0) ? bf_in : (_Float16*)nullptr);
  }
}
